// Model_32804960207401
// MI455X (gfx1250) — compile-verified, matrix operand order checked by reading, not run on hardware
//
#include <hip/hip_runtime.h>
#include <hip/hip_bf16.h>

typedef _Float16 h16v __attribute__((ext_vector_type(16)));
typedef _Float16 h8v  __attribute__((ext_vector_type(8)));
typedef _Float16 h4v  __attribute__((ext_vector_type(4)));
typedef float    f8v  __attribute__((ext_vector_type(8)));
typedef float    f4v  __attribute__((ext_vector_type(4)));

#define DHID 128

// ---------------- elementwise / graph-prep kernels ----------------

__global__ void k_fill_f32(float* __restrict__ p, float v, int n) {
  int i = blockIdx.x * blockDim.x + threadIdx.x;
  if (i < n) p[i] = v;
}

__global__ void k_degree(const int* __restrict__ dst, float* __restrict__ deg, int E) {
  int e = blockIdx.x * blockDim.x + threadIdx.x;
  if (e < E)
    __hip_atomic_fetch_add(&deg[dst[e]], 1.0f, __ATOMIC_RELAXED, __HIP_MEMORY_SCOPE_AGENT);
}

__global__ void k_rsqrt_inplace(float* __restrict__ d, int n) {
  int i = blockIdx.x * blockDim.x + threadIdx.x;
  if (i < n) d[i] = rsqrtf(d[i]);
}

// n4 = number of float4 groups
__global__ void k_f32_to_f16_v4(const float* __restrict__ in, _Float16* __restrict__ out, int n4) {
  int i = blockIdx.x * blockDim.x + threadIdx.x;
  if (i < n4) {
    f4v v = *(const f4v*)(in + (size_t)i * 4);
    h4v h = {(_Float16)v.x, (_Float16)v.y, (_Float16)v.z, (_Float16)v.w};
    *(h4v*)(out + (size_t)i * 4) = h;
  }
}

// acc[i,f..f+3] = hlin[i,f..f+3] * dis[i]^2 + b[f..f+3]  (self-loop term + bias)
__global__ void k_self_init_v4(const float* __restrict__ hlin, const float* __restrict__ dis,
                               const float* __restrict__ b, float* __restrict__ acc, int n4) {
  int i = blockIdx.x * blockDim.x + threadIdx.x;
  if (i < n4) {
    int node = i >> 5;            // 32 float4 groups per 128-wide row
    int f4i  = i & 31;
    float s = dis[node];
    float sn = s * s;
    f4v h = *(const f4v*)(hlin + (size_t)i * 4);
    f4v bb = *(const f4v*)(b + (size_t)f4i * 4);
    f4v r = {h.x * sn + bb.x, h.y * sn + bb.y, h.z * sn + bb.z, h.w * sn + bb.w};
    *(f4v*)(acc + (size_t)i * 4) = r;
  }
}

// one wave per edge; lane L handles features 4L..4L+3
__global__ void k_scatter(const int* __restrict__ src, const int* __restrict__ dst,
                          const float* __restrict__ dis, const float* __restrict__ hlin,
                          float* __restrict__ acc, int E) {
  int t = blockIdx.x * blockDim.x + threadIdx.x;
  int e = t >> 5;
  if (e >= E) return;
  int lane = t & 31;
  int s = src[e], d = dst[e];
  float w = dis[s] * dis[d];
  const f4v v = *(const f4v*)(hlin + (size_t)s * DHID + lane * 4);
  float* ap = acc + (size_t)d * DHID + lane * 4;
  __hip_atomic_fetch_add(ap + 0, v.x * w, __ATOMIC_RELAXED, __HIP_MEMORY_SCOPE_AGENT);
  __hip_atomic_fetch_add(ap + 1, v.y * w, __ATOMIC_RELAXED, __HIP_MEMORY_SCOPE_AGENT);
  __hip_atomic_fetch_add(ap + 2, v.z * w, __ATOMIC_RELAXED, __HIP_MEMORY_SCOPE_AGENT);
  __hip_atomic_fetch_add(ap + 3, v.w * w, __ATOMIC_RELAXED, __HIP_MEMORY_SCOPE_AGENT);
}

__global__ void k_relu_to_f16_v4(const float* __restrict__ acc, _Float16* __restrict__ out, int n4) {
  int i = blockIdx.x * blockDim.x + threadIdx.x;
  if (i < n4) {
    f4v v = *(const f4v*)(acc + (size_t)i * 4);
    h4v h = {(_Float16)(v.x > 0.0f ? v.x : 0.0f), (_Float16)(v.y > 0.0f ? v.y : 0.0f),
             (_Float16)(v.z > 0.0f ? v.z : 0.0f), (_Float16)(v.w > 0.0f ? v.w : 0.0f)};
    *(h4v*)(out + (size_t)i * 4) = h;
  }
}

// ---------------- weight packing into WMMA B fragments ----------------
// W is [128 x ncols] f32 row-major. Fragment (ct, kb): B tile cols ct*16..+15,
// K rows kb*32..+31. Lane (half,n): element e holds W[kb*32 + (e<8?0:16) + 8*half + (e&7)][ct*16+n].
__global__ void k_pack_w(const float* __restrict__ W, _Float16* __restrict__ Wp, int ncols) {
  int wid  = (blockIdx.x * blockDim.x + threadIdx.x) >> 5;
  int lane = threadIdx.x & 31;
  int nfrag = (ncols >> 4) * 4;
  if (wid >= nfrag) return;
  int ct = wid >> 2, kb = wid & 3;
  int n = lane & 15, half = lane >> 4;
  _Float16* dstp = Wp + (size_t)(wid * 32 + lane) * 16;
#pragma unroll
  for (int e = 0; e < 16; ++e) {
    int k = kb * 32 + ((e < 8) ? 0 : 16) + half * 8 + (e & 7);
    dstp[e] = (_Float16)W[k * ncols + ct * 16 + n];
  }
}

// ---------------- WMMA GEMM: C[N x 16*NCT] = A[N x 128] @ W ----------------
// 256 threads = 8 waves per block; each wave owns 16 rows, loops over all column tiles.
template <int NCT, bool BIAS, bool OUT_F16>
__global__ void k_gemm_wmma(const _Float16* __restrict__ A, const _Float16* __restrict__ Wp,
                            const float* __restrict__ bias, float* __restrict__ Cf,
                            _Float16* __restrict__ Ch, int nrows) {
  const int lane = threadIdx.x & 31;
  const int wave = threadIdx.x >> 5;
  const int rowBase = (blockIdx.x * 8 + wave) * 16;
  if (rowBase >= nrows) return;  // wave-uniform: EXEC stays all-ones for WMMA

  const int m = lane & 15;
  const int half = lane >> 4;
  int r = rowBase + m;
  if (r >= nrows) r = nrows - 1;  // clamp tail loads; stores predicated below

  // A fragments for all 4 K-blocks (kept in VGPRs, reused across column tiles)
  h16v a[4];
  const _Float16* ap = A + (size_t)r * DHID + half * 8;
#pragma unroll
  for (int kb = 0; kb < 4; ++kb) {
    h8v lo = *(const h8v*)(ap + kb * 32);
    h8v hi = *(const h8v*)(ap + kb * 32 + 16);
    a[kb] = __builtin_shufflevector(lo, hi, 0, 1, 2, 3, 4, 5, 6, 7,
                                    8, 9, 10, 11, 12, 13, 14, 15);
  }

  const int NC = NCT * 16;
  const bool fullTile = (rowBase + 16 <= nrows);  // wave-uniform fast path (all but 1 block)
  const int row0 = rowBase + 8 * half;            // D layout: VGPR vr -> row0 + vr, col m

#pragma unroll
  for (int ct = 0; ct < NCT; ++ct) {
    f8v c = {};
#pragma unroll
    for (int kb = 0; kb < 4; ++kb) {
      h16v b = *(const h16v*)(Wp + (size_t)((ct * 4 + kb) * 32 + lane) * 16);
      c = __builtin_amdgcn_wmma_f32_16x16x32_f16(false, a[kb], false, b,
                                                 (short)0, c, false, false);
    }
    if constexpr (BIAS) {
      float bval = bias[ct * 16 + m];
#pragma unroll
      for (int vr = 0; vr < 8; ++vr) c[vr] += bval;
    }
    if (fullTile) {
      if constexpr (OUT_F16) {
        _Float16* cp = Ch + (size_t)row0 * NC + ct * 16 + m;
#pragma unroll
        for (int vr = 0; vr < 8; ++vr) cp[vr * NC] = (_Float16)c[vr];
      } else {
        float* cp = Cf + (size_t)row0 * NC + ct * 16 + m;
#pragma unroll
        for (int vr = 0; vr < 8; ++vr) cp[vr * NC] = c[vr];
      }
    } else {
#pragma unroll
      for (int vr = 0; vr < 8; ++vr) {
        int rowS = row0 + vr;
        if (rowS < nrows) {
          if constexpr (OUT_F16)
            Ch[(size_t)rowS * NC + ct * 16 + m] = (_Float16)c[vr];
          else
            Cf[(size_t)rowS * NC + ct * 16 + m] = c[vr];
        }
      }
    }
  }
}

// ---------------- host-side launch ----------------

extern "C" void kernel_launch(void* const* d_in, const int* in_sizes, int n_in,
                              void* d_out, int out_size, void* d_ws, size_t ws_size,
                              hipStream_t stream) {
  const float* x   = (const float*)d_in[0];
  const int*   ei  = (const int*)d_in[1];
  const float* W1  = (const float*)d_in[2];
  const float* b1  = (const float*)d_in[3];
  const float* W2  = (const float*)d_in[4];
  const float* b2  = (const float*)d_in[5];
  const float* W3  = (const float*)d_in[6];
  const float* b3  = (const float*)d_in[7];
  const float* Wo1 = (const float*)d_in[8];
  const float* bo1 = (const float*)d_in[9];
  const float* Wo2 = (const float*)d_in[10];
  const float* bo2 = (const float*)d_in[11];

  const int N = in_sizes[0] / DHID;      // 50000
  const int E = in_sizes[1] / 2;         // 800000
  const int DOUT = in_sizes[10] / DHID;  // 64
  const int* src = ei;
  const int* dst = ei + E;

  // workspace carve (256B aligned)
  char* w = (char*)d_ws;
  auto carve = [&](size_t bytes) -> void* {
    void* p = (void*)w;
    w += (bytes + 255) & ~(size_t)255;
    return p;
  };
  float*    dis  = (float*)carve((size_t)N * 4);
  float*    hlin = (float*)carve((size_t)N * DHID * 4);
  float*    acc  = (float*)carve((size_t)N * DHID * 4);
  _Float16* h16  = (_Float16*)carve((size_t)N * DHID * 2);
  _Float16* h16b = (_Float16*)carve((size_t)N * DHID * 2);
  _Float16* W1p  = (_Float16*)carve(32 * 32 * 16 * 2);
  _Float16* W2p  = (_Float16*)carve(32 * 32 * 16 * 2);
  _Float16* W3p  = (_Float16*)carve(32 * 32 * 16 * 2);
  _Float16* Wo1p = (_Float16*)carve(32 * 32 * 16 * 2);
  _Float16* Wo2p = (_Float16*)carve(16 * 32 * 16 * 2);

  const int T = 256;
  const int NF = N * DHID;
  const int NF4 = NF / 4;
  dim3 blk(T);

  // degree -> dis = rsqrt(1 + indeg)
  k_fill_f32<<<(N + T - 1) / T, blk, 0, stream>>>(dis, 1.0f, N);
  k_degree<<<(E + T - 1) / T, blk, 0, stream>>>(dst, dis, E);
  k_rsqrt_inplace<<<(N + T - 1) / T, blk, 0, stream>>>(dis, N);

  // activations -> f16 ; weights -> packed f16 fragments
  k_f32_to_f16_v4<<<(NF4 + T - 1) / T, blk, 0, stream>>>(x, h16, NF4);
  k_pack_w<<<4, blk, 0, stream>>>(W1, W1p, DHID);
  k_pack_w<<<4, blk, 0, stream>>>(W2, W2p, DHID);
  k_pack_w<<<4, blk, 0, stream>>>(W3, W3p, DHID);
  k_pack_w<<<4, blk, 0, stream>>>(Wo1, Wo1p, DHID);
  k_pack_w<<<2, blk, 0, stream>>>(Wo2, Wo2p, DOUT);

  const int gemmBlocks = (N + 127) / 128;       // 8 waves * 16 rows per block
  const int scatBlocks = (E * 32 + T - 1) / T;  // one wave per edge

  const _Float16* Wps[3] = {W1p, W2p, W3p};
  const float* bs[3] = {b1, b2, b3};

  for (int layer = 0; layer < 3; ++layer) {
    k_gemm_wmma<8, false, false><<<gemmBlocks, blk, 0, stream>>>(
        h16, Wps[layer], nullptr, hlin, nullptr, N);
    k_self_init_v4<<<(NF4 + T - 1) / T, blk, 0, stream>>>(hlin, dis, bs[layer], acc, NF4);
    k_scatter<<<scatBlocks, blk, 0, stream>>>(src, dst, dis, hlin, acc, E);
    k_relu_to_f16_v4<<<(NF4 + T - 1) / T, blk, 0, stream>>>(acc, h16, NF4);
  }

  // output MLP: h @ Wo1 + bo1 (f16 intermediate), then @ Wo2 + bo2 -> f32 d_out
  k_gemm_wmma<8, true, true><<<gemmBlocks, blk, 0, stream>>>(
      h16, Wo1p, bo1, nullptr, h16b, N);
  k_gemm_wmma<4, true, false><<<gemmBlocks, blk, 0, stream>>>(
      h16b, Wo2p, bo2, (float*)d_out, nullptr, N);
}
